// MultiHeadFusionAttention_87393994539288
// MI455X (gfx1250) — matrix operand order repaired and checked by reading, not compiled, not run on hardware
//
#include <hip/hip_runtime.h>
#include <hip/hip_bf16.h>

// MI455X / gfx1250, wave32. All matmuls via v_wmma_f32_16x16x32_f16.
// Bandwidth-bound problem (~800MB HBM traffic vs 43 GFLOP) -> f16 GEMM
// intermediates, fp32 accumulation, coalesced streaming of k_gate/attn.
// fc GEMM's A-tile uses the Tensor Data Mover (tensor_load_to_lds) with
// a hand-packed D# descriptor (2D tile, LDS padding for bank-friendly pitch).

#define BB 4
#define SS 1024
#define DMM 1024
#define HH 16
#define DKK 64
#define DVV 64

typedef __attribute__((ext_vector_type(16))) _Float16 v16h;
typedef __attribute__((ext_vector_type(8)))  _Float16 v8h;
typedef __attribute__((ext_vector_type(8)))  float    v8f;
typedef __attribute__((ext_vector_type(4)))  unsigned int v4u;
typedef __attribute__((ext_vector_type(8)))  int v8i;
typedef __attribute__((ext_vector_type(4)))  int v4i;

__device__ __forceinline__ v8f wmma16(v16h a, v16h b, v8f c) {
  // D = A(16x32 f16) * B(32x16 f16) + C(16x16 f32)
  return __builtin_amdgcn_wmma_f32_16x16x32_f16(
      /*neg_a=*/false, a, /*neg_b=*/false, b,
      /*c_mod=*/(short)0, c, /*reuse_a=*/false, /*reuse_b=*/false);
}

// Build a 16-half fragment from two aligned 8-half (16B) chunks.
__device__ __forceinline__ v16h frag2(const _Float16* p0, const _Float16* p1) {
  v8h a = *(const v8h*)p0;
  v8h b = *(const v8h*)p1;
  v16h r;
#pragma unroll
  for (int i = 0; i < 8; ++i) { r[i] = a[i]; r[i + 8] = b[i]; }
  return r;
}

// ---------------------------------------------------------------------------
// Kernel 1: projection GEMM  out_h(f16)[4096x1024] = X(f32)@W(f32) + bias
// 64x64 block tile, 8 waves, each wave: 16x32 (two 16x16 WMMA tiles).
// ---------------------------------------------------------------------------
__global__ void proj_kernel(const float* __restrict__ x,
                            const float* __restrict__ w,
                            const float* __restrict__ bias,
                            _Float16* __restrict__ out_h) {
  __shared__ _Float16 lds_a[64][40];   // [m][k], 16B-aligned rows (40*2B)
  __shared__ _Float16 lds_bT[64][40];  // [n][k] (transposed -> contiguous K)
  const int nBase = blockIdx.x * 64;
  const int mBase = blockIdx.y * 64;
  const int tid = threadIdx.x, wave = tid >> 5, lane = tid & 31;
  const int waveM = wave & 3, waveN = wave >> 2;
  const int off8 = (lane >= 16) ? 8 : 0;
  const int off16 = (lane >= 16) ? 16 : 0;
  v8f c0 = {}, c1 = {};

  for (int kc = 0; kc < DMM; kc += 32) {
    {  // A tile 64x32: 8 fp32 each, convert to f16
      const int rr = tid >> 2, seg = (tid & 3) * 8;
      const float* xr = x + (size_t)(mBase + rr) * DMM + kc + seg;
#pragma unroll
      for (int t = 0; t < 8; ++t) lds_a[rr][seg + t] = (_Float16)xr[t];
    }
    {  // B tile 32x64 -> store transposed [n][k]
      const int kk = tid >> 3, seg = (tid & 7) * 8;
      const float* wr = w + (size_t)(kc + kk) * DMM + nBase + seg;
#pragma unroll
      for (int t = 0; t < 8; ++t) lds_bT[seg + t][kk] = (_Float16)wr[t];
    }
    __syncthreads();
    const int m = 16 * waveM + (lane & 15);
    v16h a = frag2(&lds_a[m][off16], &lds_a[m][off16 + 8]);
    const int n0 = 32 * waveN + (lane & 15);
    v16h b0 = frag2(&lds_bT[n0][off16], &lds_bT[n0][off16 + 8]);
    v16h b1 = frag2(&lds_bT[n0 + 16][off16], &lds_bT[n0 + 16][off16 + 8]);
    c0 = wmma16(a, b0, c0);
    c1 = wmma16(a, b1, c1);
    __syncthreads();
  }
  const int rOff = (lane >= 16) ? 8 : 0;
#pragma unroll
  for (int r = 0; r < 8; ++r) {
    const int row = mBase + 16 * waveM + r + rOff;
    const int col0 = nBase + 32 * waveN + (lane & 15);
    out_h[(size_t)row * 1024 + col0] = (_Float16)(c0[r] + bias[col0]);
    out_h[(size_t)row * 1024 + col0 + 16] = (_Float16)(c1[r] + bias[col0 + 16]);
  }
}

// ---------------------------------------------------------------------------
// Kernel 2: scores + gate + mask + softmax(k) -> attn fp32 [B,H,S,S]
// One workgroup per (b,h,16-row q strip): 16x1024 scores (64KB LDS).
// Fragments gathered straight from global (16B-contiguous per lane).
// ---------------------------------------------------------------------------
__global__ void attn_kernel(const _Float16* __restrict__ qh,
                            const _Float16* __restrict__ kh,
                            const float* __restrict__ kgate,
                            const unsigned char* __restrict__ mask,
                            float* __restrict__ attn_out) {
  __shared__ float sc[16][1024];  // exactly 64KB
  const int q0 = blockIdx.x * 16, h = blockIdx.y, b = blockIdx.z;
  const int tid = threadIdx.x, wave = tid >> 5, lane = tid & 31;

  // warm k_gate strip in GL2 (global_prefetch_b8)
  __builtin_prefetch(
      kgate + (((size_t)b * HH + h) * SS + q0 + (tid >> 4)) * SS + (tid & 15) * 64,
      0, 1);

  const int m = lane & 15;
  const int off8 = (lane >= 16) ? 8 : 0;
  const int off16 = (lane >= 16) ? 16 : 0;
  const int rOff = (lane >= 16) ? 8 : 0;

  // A fragments: q rows q0..q0+15, DK=64 -> two K-chunks of 32
  const _Float16* qrow = qh + ((size_t)(b * SS + q0 + m) * HH + h) * DKK;
  v16h aF0 = frag2(qrow + off16, qrow + off16 + 8);
  v16h aF1 = frag2(qrow + 32 + off16, qrow + 32 + off16 + 8);

  const int nBase = wave * 128;  // each wave covers 128 key positions
#pragma unroll
  for (int t = 0; t < 8; ++t) {
    const int n = nBase + t * 16 + m;  // key position for this lane's column
    const _Float16* krow = kh + ((size_t)(b * SS + n) * HH + h) * DKK;
    v16h b0 = frag2(krow + off16, krow + off16 + 8);
    v16h b1 = frag2(krow + 32 + off16, krow + 32 + off16 + 8);
    v8f c = {};
    c = wmma16(aF0, b0, c);
    c = wmma16(aF1, b1, c);
    const int col = nBase + t * 16 + m;
#pragma unroll
    for (int r = 0; r < 8; ++r) sc[r + rOff][col] = c[r];
  }
  __syncthreads();

  // softmax over k: 16 threads per row, shuffle reduce within half-wave
  const int row = tid >> 4, j = tid & 15;
  const int q = q0 + row;
  const float* grow = kgate + (((size_t)b * HH + h) * SS + q) * SS;
  const unsigned char* mrow = mask + ((size_t)b * SS + q) * SS;
  const float inv_temp = 0.125f;  // 1/sqrt(64)

  float mx = -INFINITY;
  for (int c = j; c < 1024; c += 16) {
    float v = sc[row][c] * inv_temp * grow[c];
    if (mrow[c]) v = -INFINITY;
    sc[row][c] = v;
    mx = fmaxf(mx, v);
  }
#pragma unroll
  for (int o = 1; o < 16; o <<= 1) mx = fmaxf(mx, __shfl_xor(mx, o, 16));

  float ssum = 0.f;
  for (int c = j; c < 1024; c += 16) {
    float e = __expf(sc[row][c] - mx);
    sc[row][c] = e;
    ssum += e;
  }
#pragma unroll
  for (int o = 1; o < 16; o <<= 1) ssum += __shfl_xor(ssum, o, 16);

  const float inv = 1.0f / ssum;
  float* arow = attn_out + (((size_t)b * HH + h) * SS + q) * SS;
  for (int c = j; c < 1024; c += 16) arow[c] = sc[row][c] * inv;
}

// ---------------------------------------------------------------------------
// Kernel 3: out2[b,k,h,d] = sum_q attn[b,h,q,k] * vh[b,q,h,d]   (attn^T @ V)
// One workgroup per (b,h, 64 k-rows). q streamed in chunks of 32.
// ---------------------------------------------------------------------------
__global__ void av_kernel(const float* __restrict__ attn,
                          const _Float16* __restrict__ vh,
                          _Float16* __restrict__ out2) {
  __shared__ _Float16 aT[64][40];  // [k][q] transposed attn chunk (f16)
  __shared__ _Float16 vT[64][40];  // [d][q] transposed v chunk
  const int k0 = blockIdx.x * 64, h = blockIdx.y, b = blockIdx.z;
  const int tid = threadIdx.x, wave = tid >> 5, lane = tid & 31;
  const int waveM = wave & 3, waveN = wave >> 2;
  const int off8 = (lane >= 16) ? 8 : 0;
  const int off16 = (lane >= 16) ? 16 : 0;
  v8f c0 = {}, c1 = {};
  const int qq = tid >> 3, seg = (tid & 7) * 8;

  for (int qc = 0; qc < SS; qc += 32) {
    const float* ar =
        attn + (((size_t)b * HH + h) * SS + qc + qq) * SS + k0 + seg;
    const _Float16* vr = vh + ((size_t)(b * SS + qc + qq) * HH + h) * DVV + seg;
    v8h vv = *(const v8h*)vr;
#pragma unroll
    for (int t = 0; t < 8; ++t) {
      aT[seg + t][qq] = (_Float16)ar[t];
      vT[seg + t][qq] = vv[t];
    }
    __syncthreads();
    const int mm = 16 * waveM + (lane & 15);
    v16h a = frag2(&aT[mm][off16], &aT[mm][off16 + 8]);
    const int n0 = 32 * waveN + (lane & 15);
    v16h b0 = frag2(&vT[n0][off16], &vT[n0][off16 + 8]);
    v16h b1 = frag2(&vT[n0 + 16][off16], &vT[n0 + 16][off16 + 8]);
    c0 = wmma16(a, b0, c0);
    c1 = wmma16(a, b1, c1);
    __syncthreads();
  }
  const int rOff = (lane >= 16) ? 8 : 0;
#pragma unroll
  for (int r = 0; r < 8; ++r) {
    const int kp = k0 + 16 * waveM + r + rOff;
    const int d0 = 32 * waveN + (lane & 15);
    out2[((size_t)b * SS + kp) * 1024 + h * DVV + d0] = (_Float16)c0[r];
    out2[((size_t)b * SS + kp) * 1024 + h * DVV + d0 + 16] = (_Float16)c1[r];
  }
}

// ---------------------------------------------------------------------------
// Kernel 4: y = out2 @ fc_w + fc_b; y *= struct_gate; y += residual(k)
// fp32 result into d_out (LayerNorm applied in-place by kernel 5).
// A-tile (f16, no conversion needed) is fetched by the Tensor Data Mover:
// one tensor_load_to_lds per K-step, issued by wave 0, with D#-descriptor
// LDS padding (16 DWORDs interval, 4 DWORDs pad) producing the 40-half pitch.
// ---------------------------------------------------------------------------
__global__ void fc_kernel(const _Float16* __restrict__ x,
                          const float* __restrict__ w,
                          const float* __restrict__ bias,
                          const float* __restrict__ sgate,
                          const float* __restrict__ resid,
                          float* __restrict__ out) {
  __shared__ _Float16 lds_a[64][40];
  __shared__ _Float16 lds_bT[64][40];
  const int nBase = blockIdx.x * 64;
  const int mBase = blockIdx.y * 64;
  const int tid = threadIdx.x, wave = tid >> 5, lane = tid & 31;
  const int waveM = wave & 3, waveN = wave >> 2;
  const int off8 = (lane >= 16) ? 8 : 0;
  const int off16 = (lane >= 16) ? 16 : 0;
  v8f c0 = {}, c1 = {};

  // LDS byte offset of lds_a (flat shared addr low 32 bits == LDS offset)
  const unsigned lds_a_base = (unsigned)(uintptr_t)&lds_a[0][0];

  for (int kc = 0; kc < DMM; kc += 32) {
    if (wave == 0) {
      // TDM D#: 2D tile 64(rows) x 32(halves), data_size=2B, row stride 1024,
      // pad 4 DWORDs after every 16 DWORDs (64B row -> 80B LDS pitch).
      const unsigned long long gaddr =
          (unsigned long long)(uintptr_t)(x + (size_t)mBase * DMM + kc);
      v4u g0;
      g0[0] = 1u;  // count=1 (valid), user mode, no gather
      g0[1] = lds_a_base;
      g0[2] = (unsigned)(gaddr & 0xffffffffu);
      g0[3] = (unsigned)((gaddr >> 32) & 0x1ffffffu) | (2u << 30);  // type=2
      v8i g1;
      g1[0] = (int)((1u << 16) |   // data_size = 1 -> 2 bytes
                    (1u << 20) |   // pad_enable
                    (3u << 22) |   // pad_interval: 16 DWORDs
                    (3u << 25));   // pad_amount: 4 DWORDs
      g1[1] = (int)(1024u << 16);  // tensor_dim0 = 1024 (bits 79:48 lo half)
      g1[2] = (int)(4096u << 16);  // tensor_dim1 = 4096 (bits 111:80 lo half)
      g1[3] = (int)(32u << 16);    // tile_dim0 = 32 (bits 127:112)
      g1[4] = 64;                  // tile_dim1 = 64, tile_dim2 = 0
      g1[5] = 1024;                // tensor_dim0_stride = 1024
      g1[6] = 0;
      g1[7] = 0;
      v4i gz4 = {0, 0, 0, 0};
      v8i gz8 = {0, 0, 0, 0, 0, 0, 0, 0};
      // 6-arg form (amdgpu-toolchain / clang-23 signature)
      __builtin_amdgcn_tensor_load_to_lds(g0, g1, gz4, gz4, gz8, 0);
      __builtin_amdgcn_s_wait_tensorcnt(0);
    }
    {  // B tile 32x64 fp32 -> f16, store transposed [n][k]
      const int kk = tid >> 3, seg = (tid & 7) * 8;
      const float* wr = w + (size_t)(kc + kk) * DMM + nBase + seg;
#pragma unroll
      for (int t = 0; t < 8; ++t) lds_bT[seg + t][kk] = (_Float16)wr[t];
    }
    __syncthreads();
    const int m = 16 * waveM + (lane & 15);
    v16h a = frag2(&lds_a[m][off16], &lds_a[m][off16 + 8]);
    const int n0 = 32 * waveN + (lane & 15);
    v16h b0 = frag2(&lds_bT[n0][off16], &lds_bT[n0][off16 + 8]);
    v16h b1 = frag2(&lds_bT[n0 + 16][off16], &lds_bT[n0 + 16][off16 + 8]);
    c0 = wmma16(a, b0, c0);
    c1 = wmma16(a, b1, c1);
    __syncthreads();
  }
  const int rOff = (lane >= 16) ? 8 : 0;
#pragma unroll
  for (int r = 0; r < 8; ++r) {
    const int row = mBase + 16 * waveM + r + rOff;
    const int col0 = nBase + 32 * waveN + (lane & 15);
    const float sg = sgate[row];
    float y0 = (c0[r] + bias[col0]) * sg + resid[(size_t)row * DMM + col0];
    float y1 = (c1[r] + bias[col0 + 16]) * sg + resid[(size_t)row * DMM + col0 + 16];
    out[(size_t)row * DMM + col0] = y0;
    out[(size_t)row * DMM + col0 + 16] = y1;
  }
}

// ---------------------------------------------------------------------------
// Kernel 5: in-place LayerNorm over DM=1024 per row.
// ---------------------------------------------------------------------------
__global__ void ln_kernel(float* __restrict__ out,
                          const float* __restrict__ g,
                          const float* __restrict__ bb) {
  __shared__ float s1[256], s2[256];
  const int tid = threadIdx.x;
  float* p = out + (size_t)blockIdx.x * DMM;
  float x[4], ls = 0.f, lq = 0.f;
#pragma unroll
  for (int i = 0; i < 4; ++i) {
    x[i] = p[tid + 256 * i];
    ls += x[i];
    lq += x[i] * x[i];
  }
  s1[tid] = ls;
  s2[tid] = lq;
  __syncthreads();
  for (int o = 128; o > 0; o >>= 1) {
    if (tid < o) { s1[tid] += s1[tid + o]; s2[tid] += s2[tid + o]; }
    __syncthreads();
  }
  const float mu = s1[0] * (1.0f / 1024.0f);
  const float var = s2[0] * (1.0f / 1024.0f) - mu * mu;
  const float rs = rsqrtf(var + 1e-5f);
#pragma unroll
  for (int i = 0; i < 4; ++i) {
    const int c = tid + 256 * i;
    p[c] = (x[i] - mu) * rs * g[c] + bb[c];
  }
}

// ---------------------------------------------------------------------------
extern "C" void kernel_launch(void* const* d_in, const int* in_sizes, int n_in,
                              void* d_out, int out_size, void* d_ws,
                              size_t ws_size, hipStream_t stream) {
  const float* q = (const float*)d_in[0];
  const float* k = (const float*)d_in[1];
  const float* v = (const float*)d_in[2];
  const float* kgate = (const float*)d_in[3];
  const float* sgate = (const float*)d_in[4];
  const unsigned char* mask = (const unsigned char*)d_in[5];  // jnp bool_ = 1B
  const float* wq = (const float*)d_in[6];
  const float* bq = (const float*)d_in[7];
  const float* wk = (const float*)d_in[8];
  const float* bk = (const float*)d_in[9];
  const float* wv = (const float*)d_in[10];
  const float* bv = (const float*)d_in[11];
  const float* fc_w = (const float*)d_in[12];
  const float* fc_b = (const float*)d_in[13];
  const float* ln_g = (const float*)d_in[14];
  const float* ln_b = (const float*)d_in[15];

  // d_out: out [B,S,DM] fp32 (4M floats) then attn [B,H,S,S] fp32 (64M floats)
  float* outp = (float*)d_out;
  float* attnp = outp + (size_t)BB * SS * DMM;

  // workspace: f16 qh | kh | vh | out2, 4M halves each (32MB total)
  _Float16* ws = (_Float16*)d_ws;
  const size_t SEG = (size_t)BB * SS * 1024;
  _Float16* qh = ws;
  _Float16* kh = ws + SEG;
  _Float16* vh = ws + 2 * SEG;
  _Float16* o2 = ws + 3 * SEG;

  dim3 blk(256);
  proj_kernel<<<dim3(16, 64), blk, 0, stream>>>(q, wq, bq, qh);
  proj_kernel<<<dim3(16, 64), blk, 0, stream>>>(k, wk, bk, kh);
  proj_kernel<<<dim3(16, 64), blk, 0, stream>>>(v, wv, bv, vh);
  attn_kernel<<<dim3(64, 16, 4), blk, 0, stream>>>(qh, kh, kgate, mask, attnp);
  av_kernel<<<dim3(16, 16, 4), blk, 0, stream>>>(attnp, vh, o2);
  fc_kernel<<<dim3(16, 64), blk, 0, stream>>>(o2, fc_w, fc_b, sgate, k, outp);
  ln_kernel<<<4096, blk, 0, stream>>>(outp, ln_g, ln_b);
}
